// RecursiveNN_38371237822586
// MI455X (gfx1250) — matrix operand order repaired and checked by reading, not compiled, not run on hardware
//
#include <hip/hip_runtime.h>
#include <stdint.h>

// Problem constants (match reference)
#define BATCH  128
#define DEPTH  10
#define LEAVES 1024
#define EDIM   300
#define EPAD   304            // E padded to multiple of 16
#define KDIM   608            // 2*EPAD, padded K for the pair GEMM
#define KSTEPS (KDIM / 32)    // 19 WMMA K-steps
#define CDIM   5
#define NODES  2047           // 2*LEAVES - 1

// Async global->LDS path (ASYNCcnt-tracked) if the toolchain exposes it.
#if __has_builtin(__builtin_amdgcn_global_load_async_to_lds_b64) && \
    __has_builtin(__builtin_amdgcn_global_load_async_to_lds_b128) && \
    __has_builtin(__builtin_amdgcn_s_wait_asynccnt)
#define USE_ASYNC_LDS 1
typedef int i32x4 __attribute__((vector_size(16)));
typedef int i32x2 __attribute__((vector_size(8)));
typedef const __attribute__((address_space(1))) void* gas_cv;   // generic->AS1 step
typedef __attribute__((address_space(3))) void*       las_cv;   // generic->AS3 step
typedef __attribute__((address_space(1))) i32x4*      g128_t;   // builtin param0 (b128)
typedef __attribute__((address_space(1))) i32x2*      g64_t;    // builtin param0 (b64)
typedef __attribute__((address_space(3))) i32x4*      l128_t;   // builtin param1 (b128)
typedef __attribute__((address_space(3))) i32x2*      l64_t;    // builtin param1 (b64)
#else
#define USE_ASYNC_LDS 0
#endif

typedef __attribute__((ext_vector_type(16))) __bf16 v16bf;
typedef __attribute__((ext_vector_type(8)))  float  v8f;

union Frag { uint32_t u[8]; v16bf v; };

__device__ __forceinline__ uint16_t f2bf(float f) {
    uint32_t u = __float_as_uint(f);
    u += 0x7FFFu + ((u >> 16) & 1u);        // round-to-nearest-even
    return (uint16_t)(u >> 16);
}
__device__ __forceinline__ float bf2f(uint16_t b) {
    return __uint_as_float(((uint32_t)b) << 16);
}

// ---------------------------------------------------------------------------
// Pack W_w [300, 600] fp32 into Wt [304 n][608 k] bf16, zero padded.
// pairP[k] = left[k] (k<304, cols 300..303 zero) ++ right[k-304] (304..607).
// ---------------------------------------------------------------------------
__global__ void pack_w_kernel(const float* __restrict__ Ww,
                              uint16_t* __restrict__ Wt) {
    int idx = blockIdx.x * blockDim.x + threadIdx.x;
    if (idx >= EPAD * KDIM) return;
    int n = idx / KDIM;
    int k = idx - n * KDIM;
    float v = 0.f;
    if (n < EDIM) {
        if (k < EDIM)                          v = Ww[n * (2 * EDIM) + k];
        else if (k >= EPAD && k < EPAD + EDIM) v = Ww[n * (2 * EDIM) + (k - 4)];
    }
    Wt[idx] = f2bf(v);
}

// ---------------------------------------------------------------------------
// Leaf activations: h = relu(embedding[word_ids]) -> bf16, row stride EPAD.
// ---------------------------------------------------------------------------
__global__ void leaf_kernel(const int* __restrict__ wid,
                            const float* __restrict__ emb,
                            uint16_t* __restrict__ h) {
    int row = blockIdx.x;                      // 0 .. B*L-1
    int w = wid[row];
    const float* er = emb + (long)w * EDIM;
    for (int e = threadIdx.x; e < EPAD; e += blockDim.x) {
        float v = 0.f;
        if (e < EDIM) { v = er[e]; v = v > 0.f ? v : 0.f; }
        h[(long)row * EPAD + e] = f2bf(v);
    }
}

// ---------------------------------------------------------------------------
// One tree level: h_out[R, n] = relu( sum_k pairP[R, k] * Wt[n, k] + b[n] )
// Block = 256 threads = 8 waves = 2 M-tiles x 4 N-tiles -> 32 M x 64 N.
// B slab (64 cols x 608 K) staged to LDS once; A tile (32 x 32) double
// buffered and staged per K-step via async global->LDS; 1 barrier / K-step.
// ---------------------------------------------------------------------------
__global__ void __launch_bounds__(256)
level_kernel(const uint16_t* __restrict__ hin,
             const uint16_t* __restrict__ Wt,
             const float*    __restrict__ Wb,
             uint16_t*       __restrict__ hout,
             int nHalf, int shift) {
    __shared__ uint16_t ldsB[64 * KDIM];       // 77,824 B
    __shared__ uint16_t ldsA[2][32 * 32];      //  2 x 2,048 B

    const int t    = threadIdx.x;              // 0..255
    const int wv   = t >> 5;                   // wave 0..7
    const int lane = t & 31;
    const int m0   = blockIdx.x * 32;
    const int n0   = blockIdx.y * 64;

    // ---- zero-fill LDS B rows past EPAD (edge blocks only) ----
    if (n0 + 64 > EPAD) {
        for (int i = t; i < 64 * (KDIM / 8); i += 256) {
            int r = i / (KDIM / 8);
            if (n0 + r >= EPAD) {
                int c8 = (i - r * (KDIM / 8)) * 8;
                *(uint4*)&ldsB[r * KDIM + c8] = make_uint4(0u, 0u, 0u, 0u);
            }
        }
    }
    // ---- stage B slab [n0, n0+64) x KDIM into LDS ----
    for (int i = t; i < 64 * (KDIM / 8); i += 256) {
        int r  = i / (KDIM / 8);
        int c8 = (i - r * (KDIM / 8)) * 8;     // 8 bf16 = 16 B
        int n  = n0 + r;
        if (n < EPAD) {
            const uint16_t* src = Wt + (long)n * KDIM + c8;
            uint16_t* dst = &ldsB[r * KDIM + c8];
#if USE_ASYNC_LDS
            __builtin_amdgcn_global_load_async_to_lds_b128(
                (g128_t)(gas_cv)src, (l128_t)(las_cv)dst, 0, 0);
#else
            *(uint4*)dst = *(const uint4*)src;
#endif
        }
    }

    // ---- per-thread A-staging coordinates (4 bf16 per thread, 32x32 tile) --
    const int e0 = t * 4;                      // 0..1020
    const int ar = e0 >> 5;                    // tile row 0..31
    const int ac = e0 & 31;                    // tile col (mult of 4)
    const int R  = m0 + ar;                    // output row
    const int b  = R >> shift;                 // batch index
    const int j  = R & (nHalf - 1);            // node within level
    const long leftRow = (long)b * (2 * nHalf) + 2 * j;

    auto stageA = [&](int kk, int bufidx) {
        int fp = (kk << 5) + ac;               // pairP column
        const uint16_t* src = (fp < EPAD)
            ? hin + leftRow * EPAD + fp                     // left child
            : hin + (leftRow + 1) * EPAD + (fp - EPAD);     // right child
        uint16_t* dst = &ldsA[bufidx][ar * 32 + ac];
#if USE_ASYNC_LDS
        __builtin_amdgcn_global_load_async_to_lds_b64(
            (g64_t)(gas_cv)src, (l64_t)(las_cv)dst, 0, 0);
#else
        *(uint64_t*)dst = *(const uint64_t*)src;
#endif
    };

    stageA(0, 0);                               // prologue: tile 0 in flight

    v8f acc = {};
    const int amRow = ((wv >> 2) << 4) + (lane & 15);  // A row for this lane
    const int kh    = (lane >> 4) << 3;         // A lane K-half offset (0/8)
    const int nl    = ((wv & 3) << 4) + (lane & 15);   // LDS B column

    for (int kk = 0; kk < KSTEPS; ++kk) {
#if USE_ASYNC_LDS
        __builtin_amdgcn_s_wait_asynccnt(0);    // tile kk (and B on kk=0) landed
#endif
        __syncthreads();                        // visible to all waves

        // overlap: stage tile kk+1 into the other buffer while we compute.
        // Its previous readers (iter kk-1) were consumed before this barrier.
        if (kk + 1 < KSTEPS) stageA(kk + 1, (kk + 1) & 1);

        Frag a, bm;
        const uint16_t* Abuf = ldsA[kk & 1];
        #pragma unroll
        for (int v = 0; v < 8; ++v) {           // ISA bf16 A layout (16x32)
            int k = ((v >> 2) << 4) + ((v & 3) << 1) + kh;
            a.u[v] = *(const uint32_t*)(&Abuf[amRow * 32 + k]);
        }
        int kb = (kk << 5) + ((lane >> 4) << 3);
        #pragma unroll
        for (int v = 0; v < 8; ++v)             // ISA bf16 B layout (32x16)
            bm.u[v] = *(const uint32_t*)(&ldsB[nl * KDIM + kb + ((v >> 2) << 4) + ((v & 3) << 1)]);

        acc = __builtin_amdgcn_wmma_f32_16x16x32_bf16(
                  false, a.v, false, bm.v, (short)0, acc, false, false);
    }

    // ---- epilogue: bias + relu, store bf16 (keep pad cols zero) ----
    int nG = n0 + ((wv & 3) << 4) + (lane & 15);
    if (nG < EPAD) {
        float bias = (nG < EDIM) ? Wb[nG] : 0.f;
        #pragma unroll
        for (int r = 0; r < 8; ++r) {           // ISA f32 C/D layout
            int m = ((wv >> 2) << 4) + ((lane >> 4) << 3) + r;
            float v = acc[r] + bias;
            v = v > 0.f ? v : 0.f;
            if (nG >= EDIM) v = 0.f;
            hout[(long)(m0 + m) * EPAD + nG] = f2bf(v);
        }
    }
}

// ---------------------------------------------------------------------------
// Projection: out[b, outOff+j, c] = h[row] . P_w[c] + P_b[c]   (C = 5)
// ---------------------------------------------------------------------------
__global__ void proj_kernel(const uint16_t* __restrict__ h,
                            const float* __restrict__ Pw,
                            const float* __restrict__ Pb,
                            float* __restrict__ out,
                            int nNodes, int outOff, int M) {
    int tid = blockIdx.x * blockDim.x + threadIdx.x;
    if (tid >= M) return;
    int b = tid / nNodes;
    int j = tid - b * nNodes;
    float a0 = 0.f, a1 = 0.f, a2 = 0.f, a3 = 0.f, a4 = 0.f;
    const uint16_t* hr = h + (long)tid * EPAD;
    for (int e = 0; e < EDIM; ++e) {
        float hv = bf2f(hr[e]);
        a0 += hv * Pw[0 * EDIM + e];
        a1 += hv * Pw[1 * EDIM + e];
        a2 += hv * Pw[2 * EDIM + e];
        a3 += hv * Pw[3 * EDIM + e];
        a4 += hv * Pw[4 * EDIM + e];
    }
    float* o = out + ((long)b * NODES + outOff + j) * CDIM;
    o[0] = a0 + Pb[0]; o[1] = a1 + Pb[1]; o[2] = a2 + Pb[2];
    o[3] = a3 + Pb[3]; o[4] = a4 + Pb[4];
}

// ---------------------------------------------------------------------------
extern "C" void kernel_launch(void* const* d_in, const int* in_sizes, int n_in,
                              void* d_out, int out_size, void* d_ws, size_t ws_size,
                              hipStream_t stream) {
    (void)in_sizes; (void)n_in; (void)out_size; (void)ws_size;
    const int*   word_ids  = (const int*)  d_in[0];
    const float* embedding = (const float*)d_in[1];
    const float* Ww        = (const float*)d_in[2];
    const float* Wb        = (const float*)d_in[3];
    const float* Pw        = (const float*)d_in[4];
    const float* Pb        = (const float*)d_in[5];
    float* out = (float*)d_out;

    // workspace layout (bf16 buffers)
    char* ws = (char*)d_ws;
    uint16_t* Wt = (uint16_t*)ws;                                   // 304*608*2
    uint16_t* hA = (uint16_t*)(ws + 0x60000);                       // B*L*304*2
    uint16_t* hB = (uint16_t*)(ws + 0x60000 +
                               (size_t)BATCH * LEAVES * EPAD * 2);  // half size

    { int tot = EPAD * KDIM;
      pack_w_kernel<<<(tot + 255) / 256, 256, 0, stream>>>(Ww, Wt); }

    leaf_kernel<<<BATCH * LEAVES, 128, 0, stream>>>(word_ids, embedding, hA);

    int off = 0;
    { int M = BATCH * LEAVES;
      proj_kernel<<<(M + 255) / 256, 256, 0, stream>>>(hA, Pw, Pb, out,
                                                       LEAVES, off, M);
      off += LEAVES; }

    uint16_t* hin = hA; uint16_t* hout = hB;
    for (int lvl = 1; lvl <= DEPTH; ++lvl) {
        int nHalf = LEAVES >> lvl;
        int shift = DEPTH - lvl;                // log2(nHalf)
        int M = BATCH * nHalf;                  // >= 128, multiple of 32
        level_kernel<<<dim3(M / 32, (EPAD + 63) / 64), 256, 0, stream>>>(
            hin, Wt, Wb, hout, nHalf, shift);
        proj_kernel<<<(M + 255) / 256, 256, 0, stream>>>(hout, Pw, Pb, out,
                                                         nHalf, off, M);
        off += nHalf;
        uint16_t* tmp = hin; hin = hout; hout = tmp;
    }
}
